// ComplexPatternsNet_11338713661864
// MI455X (gfx1250) — matrix operand order repaired and checked by reading, not compiled, not run on hardware
//
#include <hip/hip_runtime.h>

// ---------------------------------------------------------------------------
// CDNA5 (gfx1250) wave32 WMMA complex-GEMM pipeline for ComplexPatternsNet.
// Matrix math runs on v_wmma_f32_16x16x32_f16 (f16 in, f32 accumulate).
// Each wave computes a 16x32 complex output tile (two 16x16 WMMA N-tiles
// sharing one A fragment) to halve A-plane L2 traffic.
// ---------------------------------------------------------------------------

typedef _Float16 v16h __attribute__((ext_vector_type(16)));
typedef float    v8f  __attribute__((ext_vector_type(8)));

union Frag32B { uint4 q[2]; v16h h; };

__device__ __forceinline__ unsigned int pack_h2(float a, float b) {
  union { _Float16 h[2]; unsigned int u; } t;
  t.h[0] = (_Float16)a; t.h[1] = (_Float16)b;
  return t.u;
}

// --------------------------- small utility kernels -------------------------

__global__ void zero_kernel(float* p, int n) {
  int i = blockIdx.x * 256 + threadIdx.x;
  if (i < n) p[i] = 0.0f;
}

// Embedding gather + RoPE (complex multiply). One block per token.
__global__ __launch_bounds__(256)
void embed_rope_kernel(const int* __restrict__ tokens, const float* __restrict__ emb,
                       float* __restrict__ cRf, float* __restrict__ cIf,
                       _Float16* __restrict__ cRh, _Float16* __restrict__ cIh) {
  const int t  = blockIdx.x;          // 0..8191 (= b*2048 + l)
  const int l  = t & 2047;
  const int dc = threadIdx.x;         // 0..255
  const int tok = tokens[t];
  const float x0 = emb[(size_t)tok * 512 + 2 * dc];
  const float x1 = emb[(size_t)tok * 512 + 2 * dc + 1];
  // freq = 10000^(-dc/256) = exp(-dc * ln(10000)/256)
  const float freq = __expf(-(float)dc * (9.210340371976184f / 256.0f));
  const float ang = (float)l * freq;
  float sv, cv;
  __sincosf(ang, &sv, &cv);
  const float r = x0 * cv - x1 * sv;
  const float i = x0 * sv + x1 * cv;
  const size_t o = (size_t)t * 256 + dc;
  cRf[o] = r;  cIf[o] = i;
  cRh[o] = (_Float16)r;  cIh[o] = (_Float16)i;
}

// ------------------------- complex WMMA GEMM kernel ------------------------
// C[M,256] = A[M,256] @ W[256,256] + bias   (all complex)
// A in f16 planes (Re/Im), W f32 interleaved complex, accum f32.
// Block: 8 waves, each wave = 16 rows x 32 cols. Block tile = 128 x 32.
// MODE 0: write hl (f32 + f16 planes)
// MODE 1: sp = C; h = hl * sigmoid(|sp - hl|); write h (f32 planes)
// MODE 2: pen = C; contrib[b] += pen.mean over rows (atomic f32)
template <int MODE>
__global__ __launch_bounds__(256)
void cgemm_kernel(const _Float16* __restrict__ Arh, const _Float16* __restrict__ Aih,
                  const float* __restrict__ W, const float* __restrict__ bias,
                  const float* __restrict__ hlRf, const float* __restrict__ hlIf,
                  float* __restrict__ outRf, float* __restrict__ outIf,
                  _Float16* __restrict__ outRh, _Float16* __restrict__ outIh,
                  float* __restrict__ contribR, float* __restrict__ contribI) {
  // W slab for this 32-wide N tile, packed as f16 K-pairs, transposed
  // (n-major) with pad 132 (stride%64 = 4 u32 -> conflict-free b128 reads).
  __shared__ unsigned int lds_wr[32][132];
  __shared__ unsigned int lds_wi[32][132];

  const int tid = threadIdx.x;
  const int n0  = blockIdx.y * 32;

  for (int idx = tid; idx < 4096; idx += 256) {   // 128 K-pairs x 32 n
    const int kp = idx >> 5;
    const int nl = idx & 31;
    const float* w0 = W + ((size_t)(2 * kp) * 256 + (n0 + nl)) * 2;
    const float* w1 = w0 + 512;                   // next k row
    lds_wr[nl][kp] = pack_h2(w0[0], w1[0]);
    lds_wi[nl][kp] = pack_h2(w0[1], w1[1]);
  }
  __syncthreads();

  const int wave = tid >> 5;
  const int lane = tid & 31;
  const int hf   = lane >> 4;        // half-wave select
  const int ln   = lane & 15;
  const int m0   = blockIdx.x * 128 + wave * 16;
  const int m    = m0 + ln;

  const uint4* arR = (const uint4*)(Arh + (size_t)m * 256);
  const uint4* arI = (const uint4*)(Aih + (size_t)m * 256);

  const v8f vz = {0.f,0.f,0.f,0.f,0.f,0.f,0.f,0.f};
  v8f accRR[2] = {vz, vz}, accII[2] = {vz, vz};
  v8f accRI[2] = {vz, vz}, accIR[2] = {vz, vz};

#pragma unroll
  for (int c = 0; c < 8; ++c) {                   // K chunks of 32
    Frag32B ar, ai;
    // A fragment (contiguous k order, same as B): lanes<16 take u32 [16c..+7], lanes>=16 take [16c+8..+15]
    ar.q[0] = arR[4 * c + 2 * hf];  ar.q[1] = arR[4 * c + 2 * hf + 1];
    ai.q[0] = arI[4 * c + 2 * hf];  ai.q[1] = arI[4 * c + 2 * hf + 1];
    const int base = 16 * c + 8 * hf;
#pragma unroll
    for (int t2 = 0; t2 < 2; ++t2) {
      // B fragment: VGPR j holds K pair 16c + 8*hf + j at column (t2*16+ln)
      const unsigned int* wrRow = &lds_wr[t2 * 16 + ln][0];
      const unsigned int* wiRow = &lds_wi[t2 * 16 + ln][0];
      Frag32B br, bi;
      br.q[0] = *(const uint4*)(wrRow + base);  br.q[1] = *(const uint4*)(wrRow + base + 4);
      bi.q[0] = *(const uint4*)(wiRow + base);  bi.q[1] = *(const uint4*)(wiRow + base + 4);

      accRR[t2] = __builtin_amdgcn_wmma_f32_16x16x32_f16(false, ar.h, false, br.h, (short)0, accRR[t2], false, false);
      accII[t2] = __builtin_amdgcn_wmma_f32_16x16x32_f16(false, ai.h, false, bi.h, (short)0, accII[t2], false, false);
      accRI[t2] = __builtin_amdgcn_wmma_f32_16x16x32_f16(false, ar.h, false, bi.h, (short)0, accRI[t2], false, false);
      accIR[t2] = __builtin_amdgcn_wmma_f32_16x16x32_f16(false, ai.h, false, br.h, (short)0, accIR[t2], false, false);
    }
  }

  const int mo = m0 + hf * 8;                     // C layout: VGPR r -> M = r + 8*hf

#pragma unroll
  for (int t2 = 0; t2 < 2; ++t2) {
    const int n = n0 + t2 * 16 + ln;
    const float bR = bias[2 * n];
    const float bI = bias[2 * n + 1];

    if constexpr (MODE == 2) {
      // pen-reduce: sum the 16x16 tile over rows, accumulate contrib mean.
      float sR = 0.f, sI = 0.f;
#pragma unroll
      for (int r = 0; r < 8; ++r) {
        sR += accRR[t2][r] - accII[t2][r] + bR;
        sI += accRI[t2][r] + accIR[t2][r] + bI;
      }
      sR += __shfl_xor(sR, 16, 32);
      sI += __shfl_xor(sI, 16, 32);
      if (hf == 0) {
        const int b = m0 >> 11;                   // 2048 rows per batch
        atomicAdd(&contribR[b * 256 + n], sR * (1.0f / 2048.0f));
        atomicAdd(&contribI[b * 256 + n], sI * (1.0f / 2048.0f));
      }
    } else {
#pragma unroll
      for (int r = 0; r < 8; ++r) {
        const size_t o = (size_t)(mo + r) * 256 + n;
        float re = accRR[t2][r] - accII[t2][r] + bR;
        float im = accRI[t2][r] + accIR[t2][r] + bI;
        if constexpr (MODE == 1) {
          const float lr = hlRf[o], li = hlIf[o];
          const float dr = re - lr, di = im - li;
          const float mag = sqrtf(dr * dr + di * di);
          const float s = 1.0f / (1.0f + __expf(-mag));
          re = lr * s;  im = li * s;
          outRf[o] = re;  outIf[o] = im;
        } else {
          outRf[o] = re;  outIf[o] = im;
          outRh[o] = (_Float16)re;  outIh[o] = (_Float16)im;
        }
      }
    }
  }
}

// ----------------------- per-token pattern attention -----------------------
__global__ __launch_bounds__(256)
void attn_kernel(const float* __restrict__ hR, const float* __restrict__ hI,
                 const float* __restrict__ P,
                 float* __restrict__ mixRf, float* __restrict__ mixIf,
                 _Float16* __restrict__ mixRh, _Float16* __restrict__ mixIh) {
  const int t  = blockIdx.x;
  const int dc = threadIdx.x;
  const size_t o = (size_t)t * 256 + dc;
  const float hr = hR[o], hi = hI[o];

  float part[8];
#pragma unroll
  for (int p = 0; p < 8; ++p) {
    const float pr = P[(p * 256 + dc) * 2];
    const float pi = P[(p * 256 + dc) * 2 + 1];
    part[p] = hr * pr + hi * pi;                  // Re(h * conj(P))
  }
#pragma unroll
  for (int p = 0; p < 8; ++p)
#pragma unroll
    for (int off = 16; off > 0; off >>= 1)
      part[p] += __shfl_xor(part[p], off, 32);

  __shared__ float wred[8][8];
  __shared__ float sc[8];
  const int wave = dc >> 5, lane = dc & 31;
  if (lane == 0)
#pragma unroll
    for (int p = 0; p < 8; ++p) wred[wave][p] = part[p];
  __syncthreads();
  if (dc < 8) {
    float s = 0.f;
#pragma unroll
    for (int w = 0; w < 8; ++w) s += wred[w][dc];
    sc[dc] = s * 0.0625f;                         // 256^-0.5
  }
  __syncthreads();

  float mx = sc[0];
#pragma unroll
  for (int p = 1; p < 8; ++p) mx = fmaxf(mx, sc[p]);
  float e[8], sum = 0.f;
#pragma unroll
  for (int p = 0; p < 8; ++p) { e[p] = __expf(sc[p] - mx); sum += e[p]; }
  const float inv = 1.0f / sum;

  float ar = 0.f, ai = 0.f;
#pragma unroll
  for (int p = 0; p < 8; ++p) {
    const float a = e[p] * inv;
    ar += a * P[(p * 256 + dc) * 2];
    ai += a * P[(p * 256 + dc) * 2 + 1];
  }
  const float mr = 0.7f * hr + 0.3f * ar;
  const float mi = 0.7f * hi + 0.3f * ai;
  mixRf[o] = mr;  mixIf[o] = mi;
  mixRh[o] = (_Float16)mr;  mixIh[o] = (_Float16)mi;
}

// ------------------------- pin = contrib + mean(cur) -----------------------
__global__ __launch_bounds__(256)
void pin_kernel(const float* __restrict__ mR, const float* __restrict__ mI,
                const float* __restrict__ cR, const float* __restrict__ cI,
                float* __restrict__ pinR, float* __restrict__ pinI) {
  const int b = blockIdx.x, dc = threadIdx.x;
  const float* pR = mR + (size_t)b * 2048 * 256 + dc;
  const float* pI = mI + (size_t)b * 2048 * 256 + dc;
  float sR = 0.f, sI = 0.f;
  for (int l = 0; l < 2048; ++l) { sR += pR[(size_t)l * 256]; sI += pI[(size_t)l * 256]; }
  pinR[b * 256 + dc] = cR[b * 256 + dc] + sR * (1.0f / 2048.0f);
  pinI[b * 256 + dc] = cI[b * 256 + dc] + sI * (1.0f / 2048.0f);
}

// -------------------------------- pooled head ------------------------------
__global__ __launch_bounds__(256)
void head_kernel(const float* __restrict__ pinR, const float* __restrict__ pinI,
                 const float* __restrict__ pw, const float* __restrict__ pb,
                 const float* __restrict__ pws, const float* __restrict__ pbs,
                 const float* __restrict__ PP, float* __restrict__ feats) {
  __shared__ float pr[4][256], pi4[4][256];
  __shared__ float hlr[4][256], hli[4][256];
  __shared__ float wred[8][32];
  __shared__ float sc[32];
  const int dc = threadIdx.x;

  for (int idx = dc; idx < 1024; idx += 256) {
    pr[idx >> 8][idx & 255]  = pinR[idx];
    pi4[idx >> 8][idx & 255] = pinI[idx];
  }
  __syncthreads();

  // stage 1: hl = pin @ pw + pb
  for (int b = 0; b < 4; ++b) {
    float accr = pb[2 * dc], acci = pb[2 * dc + 1];
    for (int k = 0; k < 256; ++k) {
      const float wr = pw[(k * 256 + dc) * 2], wi = pw[(k * 256 + dc) * 2 + 1];
      const float xr = pr[b][k], xi = pi4[b][k];
      accr += xr * wr - xi * wi;
      acci += xr * wi + xi * wr;
    }
    hlr[b][dc] = accr;  hli[b][dc] = acci;
  }
  __syncthreads();

  // stage 2: sp = hl @ pws + pbs ; h = hl * sigmoid(|sp - hl|)
  float hrv[4], hiv[4];
  for (int b = 0; b < 4; ++b) {
    float accr = pbs[2 * dc], acci = pbs[2 * dc + 1];
    for (int k = 0; k < 256; ++k) {
      const float wr = pws[(k * 256 + dc) * 2], wi = pws[(k * 256 + dc) * 2 + 1];
      const float xr = hlr[b][k], xi = hli[b][k];
      accr += xr * wr - xi * wi;
      acci += xr * wi + xi * wr;
    }
    const float lr = hlr[b][dc], li = hli[b][dc];
    const float dr = accr - lr, di = acci - li;
    const float s = 1.0f / (1.0f + __expf(-sqrtf(dr * dr + di * di)));
    hrv[b] = lr * s;  hiv[b] = li * s;
  }

  // stage 3: attention over 8 pooled patterns
  float part[32];
#pragma unroll
  for (int b = 0; b < 4; ++b)
#pragma unroll
    for (int p = 0; p < 8; ++p)
      part[b * 8 + p] = hrv[b] * PP[(p * 256 + dc) * 2] + hiv[b] * PP[(p * 256 + dc) * 2 + 1];
#pragma unroll
  for (int j = 0; j < 32; ++j)
#pragma unroll
    for (int off = 16; off > 0; off >>= 1)
      part[j] += __shfl_xor(part[j], off, 32);
  const int wave = dc >> 5, lane = dc & 31;
  if (lane == 0)
#pragma unroll
    for (int j = 0; j < 32; ++j) wred[wave][j] = part[j];
  __syncthreads();
  if (dc < 32) {
    float s = 0.f;
#pragma unroll
    for (int w = 0; w < 8; ++w) s += wred[w][dc];
    sc[dc] = s * 0.0625f;
  }
  __syncthreads();

  for (int b = 0; b < 4; ++b) {
    float mx = sc[b * 8];
#pragma unroll
    for (int p = 1; p < 8; ++p) mx = fmaxf(mx, sc[b * 8 + p]);
    float e[8], sum = 0.f;
#pragma unroll
    for (int p = 0; p < 8; ++p) { e[p] = __expf(sc[b * 8 + p] - mx); sum += e[p]; }
    const float inv = 1.0f / sum;
    float ar = 0.f, ai = 0.f;
#pragma unroll
    for (int p = 0; p < 8; ++p) {
      const float a = e[p] * inv;
      ar += a * PP[(p * 256 + dc) * 2];
      ai += a * PP[(p * 256 + dc) * 2 + 1];
    }
    feats[b * 512 + 2 * dc]     = 0.7f * hrv[b] + 0.3f * ar;
    feats[b * 512 + 2 * dc + 1] = 0.7f * hiv[b] + 0.3f * ai;
  }
}

// ----------------------------- vocab projection ----------------------------
__global__ __launch_bounds__(256)
void proj_kernel(const float* __restrict__ feats, const float* __restrict__ W,
                 const float* __restrict__ bias, float* __restrict__ out) {
  __shared__ float f[2048];
  for (int idx = threadIdx.x; idx < 2048; idx += 256) f[idx] = feats[idx];
  __syncthreads();
  const int v = blockIdx.x * 256 + threadIdx.x;
  if (v >= 50257) return;
  float a0 = bias[v], a1 = a0, a2 = a0, a3 = a0;
  for (int d = 0; d < 512; ++d) {
    const float w = W[(size_t)d * 50257 + v];
    a0 += f[d] * w;
    a1 += f[512 + d] * w;
    a2 += f[1024 + d] * w;
    a3 += f[1536 + d] * w;
  }
  out[v]              = a0;
  out[50257 + v]      = a1;
  out[2 * 50257 + v]  = a2;
  out[3 * 50257 + v]  = a3;
}

// --------------------------------- launcher --------------------------------
extern "C" void kernel_launch(void* const* d_in, const int* in_sizes, int n_in,
                              void* d_out, int out_size, void* d_ws, size_t ws_size,
                              hipStream_t stream) {
  const int*   tokens = (const int*)  d_in[0];
  const float* emb    = (const float*)d_in[1];
  const float* lwp    = (const float*)d_in[2];
  const float* lbp    = (const float*)d_in[3];
  const float* lws    = (const float*)d_in[4];
  const float* lbs    = (const float*)d_in[5];
  const float* lwn    = (const float*)d_in[6];
  const float* lbn    = (const float*)d_in[7];
  const float* pats   = (const float*)d_in[8];
  const float* pw     = (const float*)d_in[9];
  const float* pb     = (const float*)d_in[10];
  const float* pws    = (const float*)d_in[11];
  const float* pbs    = (const float*)d_in[12];
  const float* pp     = (const float*)d_in[13];
  const float* outw   = (const float*)d_in[14];
  const float* outb   = (const float*)d_in[15];
  float* out = (float*)d_out;

  // Workspace carve-out (~48 MB): 4 f32 planes + 4 f16 planes + small buffers.
  const size_t PL = (size_t)8192 * 256;
  float* cRf  = (float*)d_ws;            // cur / h (f32, Re)
  float* cIf  = cRf + PL;                // cur / h (f32, Im)
  float* hlRf = cIf + PL;                // hl (f32, Re)
  float* hlIf = hlRf + PL;               // hl (f32, Im)
  _Float16* cRh  = (_Float16*)(hlIf + PL);  // cur/mixed (f16, Re)
  _Float16* cIh  = cRh + PL;
  _Float16* hlRh = cIh + PL;             // hl (f16, Re)
  _Float16* hlIh = hlRh + PL;
  float* contribR = (float*)(hlIh + PL); // [4][256]
  float* contribI = contribR + 1024;
  float* pinR  = contribI + 1024;
  float* pinI  = pinR + 1024;
  float* feats = pinI + 1024;            // [4][512]

  zero_kernel<<<8, 256, 0, stream>>>(contribR, 2048);
  embed_rope_kernel<<<8192, 256, 0, stream>>>(tokens, emb, cRf, cIf, cRh, cIh);

  const dim3 gg(64, 8);    // 64 * 128 rows = 8192, 8 * 32 cols = 256
  for (int i = 0; i < 3; ++i) {
    const float* Wp = lwp + (size_t)i * 131072;
    const float* Bp = lbp + (size_t)i * 512;
    const float* Ws = lws + (size_t)i * 131072;
    const float* Bs = lbs + (size_t)i * 512;
    const float* Wn = lwn + (size_t)i * 131072;
    const float* Bn = lbn + (size_t)i * 512;
    const float* P  = pats + (size_t)i * 4096;

    // hl = cur @ Wp + bp
    cgemm_kernel<0><<<gg, 256, 0, stream>>>(cRh, cIh, Wp, Bp, nullptr, nullptr,
                                            hlRf, hlIf, hlRh, hlIh, nullptr, nullptr);
    // sp = hl @ Ws + bs ; h = hl * sigmoid(|sp - hl|)
    cgemm_kernel<1><<<gg, 256, 0, stream>>>(hlRh, hlIh, Ws, Bs, hlRf, hlIf,
                                            cRf, cIf, nullptr, nullptr, nullptr, nullptr);
    // mixed = 0.7 h + 0.3 att(h)   (in-place on cur planes, + f16 copy)
    attn_kernel<<<8192, 256, 0, stream>>>(cRf, cIf, P, cRf, cIf, cRh, cIh);
    // contrib += mean_L(mixed @ Wn + bn)
    cgemm_kernel<2><<<gg, 256, 0, stream>>>(cRh, cIh, Wn, Bn, nullptr, nullptr,
                                            nullptr, nullptr, nullptr, nullptr,
                                            contribR, contribI);
  }

  pin_kernel<<<4, 256, 0, stream>>>(cRf, cIf, contribR, contribI, pinR, pinI);
  head_kernel<<<1, 256, 0, stream>>>(pinR, pinI, pw, pb, pws, pbs, pp, feats);
  proj_kernel<<<197, 256, 0, stream>>>(feats, outw, outb, out);
}
